// GCNConv_87935160418607
// MI455X (gfx1250) — matrix operand order repaired and checked by reading, not compiled, not run on hardware
//
#include <hip/hip_runtime.h>
#include <hip/hip_bf16.h>
#include <stdint.h>

#define NROWS 8192
#define INC   512
#define OUTC  256

typedef __attribute__((ext_vector_type(16))) __bf16 v16bf;
typedef __attribute__((ext_vector_type(8)))  __bf16 v8bf;
typedef __attribute__((ext_vector_type(8)))  float  v8f;
typedef __attribute__((ext_vector_type(4)))  unsigned int v4u;
typedef __attribute__((ext_vector_type(8)))  int v8i;
typedef __attribute__((ext_vector_type(4)))  int v4i;

__device__ __forceinline__ v8f wmma_bf16(v16bf a, v16bf b, v8f c) {
    // 8 args: (neg_a, A, neg_b, B, c_mod, C, reuse_a, reuse_b)
    return __builtin_amdgcn_wmma_f32_16x16x32_bf16(false, a, false, b, (short)0, c,
                                                   false, false);
}

// ---------------------------------------------------------------------------
// TDM: DMA one 64x32 fp32 tile of the adjacency into LDS.
// D# per CDNA5 ISA:
//   group0: count=1 | lds_addr | global_addr[56:0] | type=2
//   group1: data_size=4B, pad_enable (4 DWORDs every 32 -> LDS row stride 36
//           DWORDs => conflict-free b128 fragment reads), tensor_dim0/1=8192,
//           tile_dim0=32, tile_dim1=64, tensor_dim0_stride=8192
//   groups 2/3: zero (2-D tensor)
// ---------------------------------------------------------------------------
__device__ __forceinline__ void tdm_load_tile(unsigned int lds_byte_addr,
                                              const float* gaddr) {
    const unsigned long long ga = (unsigned long long)(uintptr_t)gaddr;
    v4u g0;
    g0[0] = 1u;                                           // count=1, user D#
    g0[1] = lds_byte_addr;                                // lds_addr
    g0[2] = (unsigned int)ga;                             // global_addr[31:0]
    g0[3] = (unsigned int)((ga >> 32) & 0x01FFFFFFu)      // global_addr[56:32]
          | 0x80000000u;                                  // type=2 ("image")
    v8i g1;
    g1[0] = (2 << 16)      // data_size = 4 bytes
          | (1 << 20)      // pad_enable
          | (4 << 22)      // pad_interval: 32 DWORDs
          | (3 << 25);     // pad_amount : 4 DWORDs
    g1[1] = (int)(8192u << 16);   // tensor_dim0[15:0]  @ bits[63:48]
    g1[2] = (int)(8192u << 16);   // tensor_dim1[15:0]  @ bits[95:80]
    g1[3] = (int)(32u << 16);     // tile_dim0 = 32     @ bits[127:112]
    g1[4] = 64;                   // tile_dim1 = 64     @ bits[143:128]
    g1[5] = 8192;                 // tensor_dim0_stride @ bits[191:160]
    g1[6] = 0;
    g1[7] = 0;
    v4i z = {0, 0, 0, 0};
#if __has_include(<hip/amd_detail/amd_gfx1250_TDM.h>)
    v8i z8 = {0, 0, 0, 0, 0, 0, 0, 0};
    __builtin_amdgcn_tensor_load_to_lds(g0, g1, z, z, z8, 0);
#else
    __builtin_amdgcn_tensor_load_to_lds(g0, g1, z, z, 0);
#endif
}

// ---------------------------------------------------------------------------
// Kernel 1: deg[i] = sum_j A[i][j];  dinv[i] = rsqrt(deg[i])
// ---------------------------------------------------------------------------
__global__ __launch_bounds__(256) void k_rowsum(const float* __restrict__ adj,
                                                float* __restrict__ dinv) {
    __shared__ float red[256];
    const int row = blockIdx.x;
    const float4* p = (const float4*)(adj + (size_t)row * NROWS);
    float s = 0.f;
    for (int j = threadIdx.x; j < NROWS / 4; j += 256) {
        float4 v = p[j];
        s += v.x + v.y + v.z + v.w;
    }
    red[threadIdx.x] = s;
    __syncthreads();
    for (int off = 128; off > 0; off >>= 1) {
        if (threadIdx.x < off) red[threadIdx.x] += red[threadIdx.x + off];
        __syncthreads();
    }
    if (threadIdx.x == 0) dinv[row] = rsqrtf(red[0]);
}

// ---------------------------------------------------------------------------
// Kernel 2: Wt_hi/Wt_lo[n][k] = split_bf16(weight[k][n])  (transpose + split)
// ---------------------------------------------------------------------------
__global__ __launch_bounds__(256) void k_wsplit(const float* __restrict__ w,
                                                __bf16* __restrict__ wthi,
                                                __bf16* __restrict__ wtlo) {
    const int g = blockIdx.x * 256 + threadIdx.x;
    const int k = g & (INC - 1);
    const int n = g >> 9;
    float x = w[(size_t)k * OUTC + n];
    __bf16 h = (__bf16)x;
    wthi[(size_t)n * INC + k] = h;
    wtlo[(size_t)n * INC + k] = (__bf16)(x - (float)h);
}

// ---------------------------------------------------------------------------
// Kernel 3: Yt_hi/Yt_lo[n][i] = split_bf16( dinv[i] * (X @ W)[i][n] )
// Split-bf16 WMMA GEMM over K=512; epilogue stores transposed + row-scaled.
// ---------------------------------------------------------------------------
__global__ __launch_bounds__(256) void k_xw(const float* __restrict__ x,
                                            const __bf16* __restrict__ wthi,
                                            const __bf16* __restrict__ wtlo,
                                            const float* __restrict__ dinv,
                                            __bf16* __restrict__ ythi,
                                            __bf16* __restrict__ ytlo) {
    const int lane  = threadIdx.x & 31;
    const int wave  = threadIdx.x >> 5;
    const int lmod  = lane & 15;
    const int lhalf = lane >> 4;
    const int m0 = blockIdx.x * 128 + wave * 16;
    const int n0 = blockIdx.y * 16;

    v8f acc = {};
    const float*  arow = x    + (size_t)(m0 + lmod) * INC;
    const __bf16* bh   = wthi + (size_t)(n0 + lmod) * INC + lhalf * 16;
    const __bf16* bl   = wtlo + (size_t)(n0 + lmod) * INC + lhalf * 16;
    const int akb = lhalf * 16;

    for (int k0 = 0; k0 < INC; k0 += 32) {
        float4 a0 = *(const float4*)(arow + k0 + akb);
        float4 a1 = *(const float4*)(arow + k0 + akb + 4);
        float4 a2 = *(const float4*)(arow + k0 + akb + 8);
        float4 a3 = *(const float4*)(arow + k0 + akb + 12);
        float af[16] = {a0.x, a0.y, a0.z, a0.w, a1.x, a1.y, a1.z, a1.w,
                        a2.x, a2.y, a2.z, a2.w, a3.x, a3.y, a3.z, a3.w};
        v16bf ahi, alo;
#pragma unroll
        for (int i = 0; i < 16; ++i) {
            __bf16 h = (__bf16)af[i];
            ahi[i] = h;
            alo[i] = (__bf16)(af[i] - (float)h);
        }
        v16bf bhv = *(const v16bf*)(bh + k0);
        v16bf blv = *(const v16bf*)(bl + k0);
        acc = wmma_bf16(ahi, bhv, acc);
        acc = wmma_bf16(alo, bhv, acc);
        acc = wmma_bf16(ahi, blv, acc);
    }

    const int i0 = m0 + lhalf * 8;
    float4 d0 = *(const float4*)(dinv + i0);
    float4 d1 = *(const float4*)(dinv + i0 + 4);
    float dv[8] = {d0.x, d0.y, d0.z, d0.w, d1.x, d1.y, d1.z, d1.w};
    v8bf oh, ol;
#pragma unroll
    for (int r = 0; r < 8; ++r) {
        float y  = dv[r] * acc[r];
        __bf16 h = (__bf16)y;
        oh[r] = h;
        ol[r] = (__bf16)(y - (float)h);
    }
    const int n = n0 + lmod;
    *(v8bf*)(ythi + (size_t)n * NROWS + i0) = oh;
    *(v8bf*)(ytlo + (size_t)n * NROWS + i0) = ol;
}

// ---------------------------------------------------------------------------
// Kernel 4: out = relu( dinv[i] * ( (A @ Y)[i][n] + Y[i][n] ) ),  Y = Dinv*XW
// TDM double-buffered: wave 0 issues tensor_load_to_lds for slab i+2 while
// all 8 waves (4M x 2N) run the 3-product split-bf16 WMMAs on slab i.
// LDS layout: fp32 tile rows at 36-DWORD stride (TDM pad) -> conflict-free.
// ---------------------------------------------------------------------------
__global__ __launch_bounds__(256) void k_agg(const float* __restrict__ adj,
                                             const __bf16* __restrict__ ythi,
                                             const __bf16* __restrict__ ytlo,
                                             const float* __restrict__ dinv,
                                             float* __restrict__ out) {
    __shared__ float sA[2][64 * 36];   // 2 x 9216 B, rows padded 32->36 DWORDs

    const int tid   = threadIdx.x;
    const int lane  = tid & 31;
    const int wave  = tid >> 5;
    const int lmod  = lane & 15;
    const int lhalf = lane >> 4;
    const int wm = wave >> 1;                 // 0..3 (M)
    const int wn = wave & 1;                  // 0..1 (N)
    const int row0 = blockIdx.x * 64;
    const int col0 = blockIdx.y * 128 + wn * 64;

    v8f acc[4] = {};

    const int a_r  = wm * 16 + lmod;          // fragment row in tile
    const int a_kb = lhalf * 16;              // fragment K chunk base (0 or 16)

    const unsigned int lds0 = (unsigned int)(uintptr_t)&sA[0][0];
    const unsigned int lds1 = (unsigned int)(uintptr_t)&sA[1][0];
    const float* abase = adj + (size_t)row0 * NROWS;

    if (wave == 0) {                          // prologue: fill both buffers
        tdm_load_tile(lds0, abase + 0);
        tdm_load_tile(lds1, abase + 32);
    }

    for (int k0 = 0; k0 < NROWS; k0 += 32) {
        const int cur = (k0 >> 5) & 1;
        if (wave == 0) __builtin_amdgcn_s_wait_tensorcnt(1);
        __syncthreads();                      // slab `cur` resident for all

        // A fragments: fp32 from padded LDS, split to bf16 hi/lo in-register
        const float* sp = &sA[cur][36 * a_r + a_kb];
        float4 c0 = *(const float4*)(sp);
        float4 c1 = *(const float4*)(sp + 4);
        float4 c2 = *(const float4*)(sp + 8);
        float4 c3 = *(const float4*)(sp + 12);
        float af[16] = {c0.x, c0.y, c0.z, c0.w, c1.x, c1.y, c1.z, c1.w,
                        c2.x, c2.y, c2.z, c2.w, c3.x, c3.y, c3.z, c3.w};
        v16bf ahi, alo;
#pragma unroll
        for (int i = 0; i < 16; ++i) {
            __bf16 h = (__bf16)af[i];
            ahi[i] = h;
            alo[i] = (__bf16)(af[i] - (float)h);
        }

#pragma unroll
        for (int nt = 0; nt < 4; ++nt) {
            const size_t bo = (size_t)(col0 + nt * 16 + lmod) * NROWS + k0 + lhalf * 16;
            v16bf bhv = *(const v16bf*)(ythi + bo);
            v16bf blv = *(const v16bf*)(ytlo + bo);
            acc[nt] = wmma_bf16(ahi, bhv, acc[nt]);
            acc[nt] = wmma_bf16(alo, bhv, acc[nt]);
            acc[nt] = wmma_bf16(ahi, blv, acc[nt]);
        }
        __syncthreads();                      // all waves done reading `cur`

        if (wave == 0 && (k0 + 64) < NROWS) { // refill `cur` for slab i+2
            tdm_load_tile(cur ? lds1 : lds0, abase + (k0 + 64));
        }
    }

    // Epilogue: + identity term Y[i][n], scale by dinv[i], ReLU, store fp32.
    const int i0 = row0 + wm * 16 + lhalf * 8;
    float4 d0 = *(const float4*)(dinv + i0);
    float4 d1 = *(const float4*)(dinv + i0 + 4);
    float dv[8] = {d0.x, d0.y, d0.z, d0.w, d1.x, d1.y, d1.z, d1.w};
#pragma unroll
    for (int nt = 0; nt < 4; ++nt) {
        const int n = col0 + nt * 16 + lmod;
        v8bf yh = *(const v8bf*)(ythi + (size_t)n * NROWS + i0);
        v8bf yl = *(const v8bf*)(ytlo + (size_t)n * NROWS + i0);
#pragma unroll
        for (int r = 0; r < 8; ++r) {
            float v = dv[r] * (acc[nt][r] + (float)yh[r] + (float)yl[r]);
            out[(size_t)(i0 + r) * OUTC + n] = v > 0.f ? v : 0.f;
        }
    }
}

// ---------------------------------------------------------------------------
extern "C" void kernel_launch(void* const* d_in, const int* in_sizes, int n_in,
                              void* d_out, int out_size, void* d_ws, size_t ws_size,
                              hipStream_t stream) {
    (void)in_sizes; (void)n_in; (void)out_size; (void)ws_size;
    const float* input  = (const float*)d_in[0];   // [N, IN_C]
    const float* adj    = (const float*)d_in[1];   // [N, N]
    const float* weight = (const float*)d_in[2];   // [IN_C, OUT_C]
    float* out = (float*)d_out;                    // [N, OUT_C]

    char* ws = (char*)d_ws;
    size_t off = 0;
    float*  dinv = (float*) (ws + off); off += (size_t)NROWS * 4;          // 32 KB
    __bf16* wthi = (__bf16*)(ws + off); off += (size_t)OUTC * INC * 2;     // 256 KB
    __bf16* wtlo = (__bf16*)(ws + off); off += (size_t)OUTC * INC * 2;     // 256 KB
    __bf16* ythi = (__bf16*)(ws + off); off += (size_t)OUTC * NROWS * 2;   // 4 MB
    __bf16* ytlo = (__bf16*)(ws + off); off += (size_t)OUTC * NROWS * 2;   // 4 MB

    k_rowsum<<<NROWS, 256, 0, stream>>>(adj, dinv);
    k_wsplit<<<(OUTC * INC) / 256, 256, 0, stream>>>(weight, wthi, wtlo);
    k_xw<<<dim3(NROWS / 128, OUTC / 16), 256, 0, stream>>>(input, wthi, wtlo, dinv,
                                                           ythi, ytlo);
    k_agg<<<dim3(NROWS / 64, OUTC / 128), 256, 0, stream>>>(adj, ythi, ytlo, dinv, out);
}
